// Decoder_arch2_38946763440490
// MI455X (gfx1250) — matrix operand order repaired and checked by reading, not compiled, not run on hardware
//
#include <hip/hip_runtime.h>
#include <hip/hip_bf16.h>
#include <stdint.h>

// Problem constants (match reference)
#define BDIM 128
#define TDIM 24
#define EDIM 512
#define VDIM 30000
#define VPAD 30016  // VDIM rounded up to 64 (full WMMA n-tiles, zero-padded B rows)
#define H2D  1024
#define KDIM 1024   // IN == H2 == 1024 (K of every GEMM)
#define G4   4096   // 4*H2

typedef __bf16 bf16_t;
typedef __bf16 bf16x8 __attribute__((ext_vector_type(8)));
typedef __bf16 v16bf  __attribute__((ext_vector_type(16)));
typedef float  v8f    __attribute__((ext_vector_type(8)));

__device__ __forceinline__ unsigned short f2bf(float f) {
  union { float f; unsigned u; } v; v.f = f;
  unsigned r = v.u + 0x7fffu + ((v.u >> 16) & 1u);   // round-to-nearest-even
  return (unsigned short)(r >> 16);
}

__device__ __forceinline__ float sigmoidf(float x) {
  return 1.0f / (1.0f + __expf(-x));
}

// ---------------------------------------------------------------------------
// f32 -> bf16 conversion (vectorized x4); indices >= n4_in write zeros so the
// destination can be zero-padded past the source (W_out 30000 -> 30016 rows).
// ---------------------------------------------------------------------------
__global__ void cvt_f32_to_bf16(const float* __restrict__ in,
                                unsigned short* __restrict__ out,
                                int n4_in, int n4_out) {
  int i = blockIdx.x * blockDim.x + threadIdx.x;
  if (i >= n4_out) return;
  float4 v;
  if (i < n4_in) v = ((const float4*)in)[i];
  else { v.x = 0.f; v.y = 0.f; v.z = 0.f; v.w = 0.f; }
  ushort4 o;
  o.x = f2bf(v.x); o.y = f2bf(v.y); o.z = f2bf(v.z); o.w = f2bf(v.w);
  ((ushort4*)out)[i] = o;
}

// ---------------------------------------------------------------------------
// Build x_bf16[m=t*128+b][k]: k<512 -> emb[token(t,b)][k], else features[b][k-512]
// token(0,b) = pad_idx; token(t,b) = captions[b][t-1]
// ---------------------------------------------------------------------------
__global__ void build_x(const float* __restrict__ emb,
                        const float* __restrict__ features,
                        const int* __restrict__ captions,
                        const int* __restrict__ pad_idx,
                        unsigned short* __restrict__ x) {
  int idx = blockIdx.x * blockDim.x + threadIdx.x;
  if (idx >= BDIM * TDIM * KDIM) return;
  int j = idx & (KDIM - 1);
  int m = idx >> 10;      // t*128 + b
  int t = m >> 7;
  int b = m & 127;
  float v;
  if (j < EDIM) {
    int tok = (t == 0) ? pad_idx[0] : captions[b * TDIM + (t - 1)];
    v = emb[(size_t)tok * EDIM + j];
  } else {
    v = features[(size_t)b * EDIM + (j - EDIM)];
  }
  x[idx] = f2bf(v);
}

// ---------------------------------------------------------------------------
// WMMA fragment loaders (CDNA5 ISA 7.12.2 layouts, wave32)
// A 16x32 bf16: lane L holds row M=L%16; elems 0..7 = K k0..k0+7, 8..15 = k0+16..k0+23,
//               k0 = kt + (L<16 ? 0 : 8)
// B 32x16 bf16: lane L holds col N=L%16; elems 0..7 = K k0..k0+7, 8..15 = k0+16..k0+23,
//               k0 = kt + (L<16 ? 0 : 8)
// ---------------------------------------------------------------------------
__device__ __forceinline__ v16bf load_a_frag(const bf16_t* p) {
  bf16x8 lo = *(const bf16x8*)(p);
  bf16x8 hi = *(const bf16x8*)(p + 16);
  v16bf r;
#pragma unroll
  for (int i = 0; i < 8; ++i) { r[i] = lo[i]; r[i + 8] = hi[i]; }
  return r;
}
__device__ __forceinline__ v16bf load_b_frag(const bf16_t* p) {
  bf16x8 lo = *(const bf16x8*)(p);
  bf16x8 hi = *(const bf16x8*)(p + 16);
  v16bf r;
#pragma unroll
  for (int i = 0; i < 8; ++i) { r[i] = lo[i]; r[i + 8] = hi[i]; }
  return r;
}

// ---------------------------------------------------------------------------
// Generic bf16 GEMM, f32 accumulate:  C[M][N] = A[M][K=1024] * B[Npad][K]^T
//   + optional cinit (f32, ld=N, same rows as C)  [recurrent xg fold-in]
//   + optional bias1/bias2 (per-n)
// TRANS_OUT: scatter into predicted[b][v][t] with m = t*128+b, n = v
// One wave computes a (16*MT)x64 tile: MT M-subtiles x 4 N-subtiles.
//   MT=3 (48x64): 12 accs = 96 VGPRs, 14 b128 loads / 12 WMMAs per K-step
//                 (~27 FLOP per L2 byte) -- fits the VGPR file, no spills
//   MT=2 (32x64): small recurrent GEMM (keeps wave count at 256)
//   (MT=4 spilled accumulators to scratch -- do not use)
// B is padded to a multiple of 64 rows, so the K loop is branch-free and fully
// unrolled (all loads are base-pointer + immediate-offset b128 clauses);
// boundary handling lives only in the store epilogue (wave-uniform skip).
// ---------------------------------------------------------------------------
template <int MT, bool HAS_CINIT, bool HAS_B1, bool HAS_B2, bool TRANS_OUT>
__global__ void __launch_bounds__(32)
gemm_bf16(const unsigned short* __restrict__ Abits,
          const unsigned short* __restrict__ Bbits,
          const float* __restrict__ cinit,
          const float* __restrict__ bias1,
          const float* __restrict__ bias2,
          float* __restrict__ out, int N) {
  const bf16_t* A = (const bf16_t*)Abits;
  const bf16_t* B = (const bf16_t*)Bbits;
  const int lane = threadIdx.x;
  const int l16  = lane & 15;
  const int hi   = lane >> 4;          // 0 or 1 (lane group)
  const int mBase = blockIdx.y * (16 * MT);
  const int nBase = blockIdx.x * 64;

  // Loop-invariant per-lane base pointers (K-group offsets folded in).
  const bf16_t* arow[MT];
#pragma unroll
  for (int i = 0; i < MT; ++i)
    arow[i] = A + (size_t)(mBase + 16 * i + l16) * KDIM + hi * 8;
  const bf16_t* brow[4];
#pragma unroll
  for (int j = 0; j < 4; ++j)
    brow[j] = B + (size_t)(nBase + 16 * j + l16) * KDIM + hi * 8;

  v8f acc[MT][4];
  const v8f vzero = {0.f, 0.f, 0.f, 0.f, 0.f, 0.f, 0.f, 0.f};
#pragma unroll
  for (int i = 0; i < MT; ++i)
#pragma unroll
    for (int j = 0; j < 4; ++j) acc[i][j] = vzero;

#pragma unroll
  for (int kt = 0; kt < KDIM; kt += 32) {
    v16bf a[MT];
#pragma unroll
    for (int i = 0; i < MT; ++i) a[i] = load_a_frag(arow[i] + kt);
#pragma unroll
    for (int j = 0; j < 4; ++j) {
      v16bf bfr = load_b_frag(brow[j] + kt);
#pragma unroll
      for (int i = 0; i < MT; ++i)
        acc[i][j] = __builtin_amdgcn_wmma_f32_16x16x32_bf16(
            false, a[i], false, bfr, (short)0, acc[i][j], false, false);
    }
  }

  // C/D layout: VGPR r of lane L -> M = r + 8*(L/16), N = L%16 (within 16x16 tile)
#pragma unroll
  for (int i = 0; i < MT; ++i) {
#pragma unroll
    for (int j = 0; j < 4; ++j) {
      int ncol = nBase + 16 * j;
      if (ncol >= N) continue;            // wave-uniform (N % 16 == 0)
      int n = ncol + l16;
      float badd = 0.0f;
      if (HAS_B1) badd += bias1[n];
      if (HAS_B2) badd += bias2[n];
#pragma unroll
      for (int r = 0; r < 8; ++r) {
        int m = mBase + 16 * i + hi * 8 + r;
        float v = acc[i][j][r] + badd;
        if (HAS_CINIT) v += cinit[(size_t)m * N + n];
        if (TRANS_OUT) {
          int t = m >> 7, b = m & 127;
          out[(size_t)b * ((size_t)VDIM * TDIM) + (size_t)n * TDIM + t] = v;
        } else {
          out[(size_t)m * N + n] = v;
        }
      }
    }
  }
}

// ---------------------------------------------------------------------------
// LSTM cell elementwise: g[128][4096] (i,f,g,o torch order) -> c (f32), h (bf16)
// ---------------------------------------------------------------------------
template <bool FIRST>
__global__ void lstm_cell(const float* __restrict__ g, float* __restrict__ c,
                          unsigned short* __restrict__ h_out) {
  int idx = blockIdx.x * blockDim.x + threadIdx.x;
  if (idx >= BDIM * H2D) return;
  int b = idx >> 10;
  int j = idx & (H2D - 1);
  const float* gr = g + (size_t)b * G4;
  float gi = sigmoidf(gr[j]);
  float gf = sigmoidf(gr[j + H2D]);
  float gg = tanhf(gr[j + 2 * H2D]);
  float go = sigmoidf(gr[j + 3 * H2D]);
  float cp = FIRST ? 0.0f : c[idx];
  float cn = gf * cp + gi * gg;
  c[idx] = cn;
  h_out[idx] = f2bf(go * tanhf(cn));
}

// ---------------------------------------------------------------------------
// Launch: ws layout (~144 MB):
//   Wih_b(8.4M) Whh_b(8.4M) Wout_b(61.5M, padded) x_b(6.3M) xg(50.3M)
//   hs_b(6.3M) c(0.5M) gbuf(2.1M)
// ---------------------------------------------------------------------------
extern "C" void kernel_launch(void* const* d_in, const int* in_sizes, int n_in,
                              void* d_out, int out_size, void* d_ws, size_t ws_size,
                              hipStream_t stream) {
  (void)in_sizes; (void)n_in; (void)out_size; (void)ws_size;
  const float* features = (const float*)d_in[0];
  const int*   captions = (const int*)d_in[1];
  const int*   pad_idx  = (const int*)d_in[2];
  const float* emb      = (const float*)d_in[3];
  const float* W_ih     = (const float*)d_in[4];
  const float* W_hh     = (const float*)d_in[5];
  const float* b_ih     = (const float*)d_in[6];
  const float* b_hh     = (const float*)d_in[7];
  const float* W_out    = (const float*)d_in[8];
  const float* b_out    = (const float*)d_in[9];
  float* out = (float*)d_out;

  char* ws = (char*)d_ws;
  size_t off = 0;
  auto alloc = [&](size_t bytes) -> char* {
    char* p = ws + off;
    off = (off + bytes + 255) & ~(size_t)255;
    return p;
  };
  unsigned short* Wih_b  = (unsigned short*)alloc((size_t)G4 * KDIM * 2);
  unsigned short* Whh_b  = (unsigned short*)alloc((size_t)G4 * KDIM * 2);
  unsigned short* Wout_b = (unsigned short*)alloc((size_t)VPAD * KDIM * 2);
  unsigned short* x_b    = (unsigned short*)alloc((size_t)BDIM * TDIM * KDIM * 2);
  float*          xg     = (float*)alloc((size_t)BDIM * TDIM * G4 * 4);
  unsigned short* hs_b   = (unsigned short*)alloc((size_t)BDIM * TDIM * H2D * 2);
  float*          cbuf   = (float*)alloc((size_t)BDIM * H2D * 4);
  float*          gbuf   = (float*)alloc((size_t)BDIM * G4 * 4);

  const int thr = 256;

  // 1) weights -> bf16 (W_out zero-padded to VPAD rows)
  {
    int n4 = G4 * KDIM / 4;
    cvt_f32_to_bf16<<<(n4 + thr - 1) / thr, thr, 0, stream>>>(W_ih, Wih_b, n4, n4);
    cvt_f32_to_bf16<<<(n4 + thr - 1) / thr, thr, 0, stream>>>(W_hh, Whh_b, n4, n4);
    int n4i = VDIM * KDIM / 4, n4o = VPAD * KDIM / 4;
    cvt_f32_to_bf16<<<(n4o + thr - 1) / thr, thr, 0, stream>>>(W_out, Wout_b, n4i, n4o);
  }

  // 2) embedding gather + concat -> bf16 A matrix (rows m = t*128 + b)
  build_x<<<(BDIM * TDIM * KDIM + thr - 1) / thr, thr, 0, stream>>>(
      emb, features, captions, pad_idx, x_b);

  // 3) xg = x @ W_ih^T + b_ih + b_hh   (M=3072, N=4096), 48x64 tiles
  {
    dim3 grid(G4 / 64, (BDIM * TDIM) / 48);
    gemm_bf16<3, false, true, true, false><<<grid, 32, 0, stream>>>(
        x_b, Wih_b, nullptr, b_ih, b_hh, xg, G4);
  }

  // 4) recurrent scan: 24 steps (h0 = 0 => step 0 skips its GEMM), 32x64 tiles
  for (int t = 0; t < TDIM; ++t) {
    const float* gsrc;
    if (t == 0) {
      gsrc = xg;                     // g = xg[0] since h_{-1} = 0
    } else {
      dim3 grid(G4 / 64, BDIM / 32);
      gemm_bf16<2, true, false, false, false><<<grid, 32, 0, stream>>>(
          hs_b + (size_t)(t - 1) * BDIM * H2D, Whh_b,
          xg + (size_t)t * BDIM * G4, nullptr, nullptr, gbuf, G4);
      gsrc = gbuf;
    }
    int blocks = (BDIM * H2D + thr - 1) / thr;
    if (t == 0)
      lstm_cell<true><<<blocks, thr, 0, stream>>>(gsrc, cbuf, hs_b);
    else
      lstm_cell<false><<<blocks, thr, 0, stream>>>(
          gsrc, cbuf, hs_b + (size_t)t * BDIM * H2D);
  }

  // 5) predicted[b][v][t] = hs @ W_out^T + b_out (M=3072, N=30000, fused transpose)
  {
    dim3 grid(VPAD / 64, (BDIM * TDIM) / 48);
    gemm_bf16<3, false, true, false, true><<<grid, 32, 0, stream>>>(
        hs_b, Wout_b, nullptr, b_out, nullptr, out, VDIM);
  }
}
